// FlashLongconvLayer_81578608820277
// MI455X (gfx1250) — matrix operand order repaired and checked by reading, not compiled, not run on hardware
//
#include <hip/hip_runtime.h>
#include <hip/hip_bf16.h>
#include <math.h>
#include <stdint.h>

// ---------------- types for WMMA (CDNA5 gfx1250, wave32) ----------------
typedef __attribute__((ext_vector_type(16))) __bf16 v16bf;
typedef __attribute__((ext_vector_type(8)))  __bf16 v8bf;
typedef __attribute__((ext_vector_type(8)))  float  v8f;

union FragB16 { v16bf v; struct { v8bf lo, hi; } p; };

// ---------------- problem constants (match reference) ----------------
constexpr int CB  = 2;      // batch
constexpr int CL  = 4096;   // sequence length
constexpr int CH  = 2048;   // hidden
constexpr int CLK = 128;    // conv kernel taps
constexpr float LAM = 0.003f;

constexpr int GM = CB * CL;   // 8192 GEMM rows
constexpr int GK = CH;        // 2048 reduction
constexpr int GN = CH;        // 2048 output cols (per GLU half)

// GEMM tiling
constexpr int BM = 128;       // block M tile
constexpr int BN = 64;        // block out-N tile (implies 128 GEMM cols: a+g)
constexpr int KC = 32;        // K step (one wmma K)
constexpr int LDSW = 40;      // bf16 units per LDS row (80B: 64B data + 16B pad)

// =====================================================================
// Kernel squash: ks = relu(|k|-lam)*sign(k)
// =====================================================================
__global__ __launch_bounds__(256) void squash_kernel(const float* __restrict__ k,
                                                     float* __restrict__ ks, int n) {
  int i = blockIdx.x * 256 + threadIdx.x;
  if (i < n) {
    float v = k[i];
    float a = fmaxf(fabsf(v) - LAM, 0.0f);
    ks[i] = copysignf(a, v);
  }
}

// =====================================================================
// W fp32 -> bf16
// =====================================================================
__global__ __launch_bounds__(256) void w2bf_kernel(const float* __restrict__ W,
                                                   __bf16* __restrict__ Wb, int n) {
  int i = blockIdx.x * 256 + threadIdx.x;
  if (i < n) Wb[i] = (__bf16)W[i];
}

// =====================================================================
// Direct causal conv (Lk=128) + 2*D*u skip + exact GELU, output bf16
// =====================================================================
constexpr int TL = 32;   // outputs per thread along l
constexpr int JC = 16;   // taps per chunk

__global__ __launch_bounds__(256) void conv_gelu_kernel(const float* __restrict__ u,
                                                        const float* __restrict__ ks,
                                                        const float* __restrict__ D,
                                                        __bf16* __restrict__ Yb) {
  const int h  = blockIdx.x * 256 + threadIdx.x;   // 0..H-1
  const int l0 = blockIdx.y * TL;
  const int b  = blockIdx.z;
  const float* __restrict__ ub = u + (size_t)b * CL * CH;
  const float* __restrict__ kh = ks + (size_t)h * CLK;

  const float dv = 2.0f * D[h];

  float acc[TL];
#pragma unroll
  for (int t = 0; t < TL; ++t)
    acc[t] = dv * ub[(size_t)(l0 + t) * CH + h];

  for (int jc = 0; jc < CLK; jc += JC) {
    float kreg[JC];
#pragma unroll
    for (int jj = 0; jj < JC; ++jj) kreg[jj] = kh[jc + jj];

    float uw[TL + JC - 1];
    const int lbase = l0 - jc - (JC - 1);
#pragma unroll
    for (int s = 0; s < TL + JC - 1; ++s) {
      int l = lbase + s;
      uw[s] = (l >= 0) ? ub[(size_t)l * CH + h] : 0.0f;
    }
#pragma unroll
    for (int jj = 0; jj < JC; ++jj)
#pragma unroll
      for (int t = 0; t < TL; ++t)
        acc[t] += kreg[jj] * uw[t + (JC - 1) - jj];
  }

  const size_t outb = ((size_t)b * CL + l0) * CH + h;
#pragma unroll
  for (int t = 0; t < TL; ++t) {
    float y = acc[t];
    float g = 0.5f * y * (1.0f + erff(y * 0.70710678118654752f));
    Yb[outb + (size_t)t * CH] = (__bf16)g;
  }
}

// =====================================================================
// bf16 WMMA GEMM with async global->LDS staging (double buffered) and
// fused bias + GLU epilogue.
// Block: 256 threads = 8 waves; tile 128(M) x 64(outN) -> 128 GEMM cols.
// Per K-step: 1 barrier; tiles move via GLOBAL_LOAD_ASYNC_TO_LDS_B128
// (ASYNCcnt), fragments via ds_load_b128, math via v_wmma (8 per step).
// =====================================================================
__device__ inline v8f vzero8() {
  v8f v;
#pragma unroll
  for (int i = 0; i < 8; ++i) v[i] = 0.0f;
  return v;
}

// per-lane async copy: 16B global -> 16B LDS (tracked by ASYNCcnt)
__device__ inline void async_b128(uint32_t lds_off, const void* gaddr) {
  asm volatile("global_load_async_to_lds_b128 %0, %1, off"
               :: "v"(lds_off), "v"((unsigned long long)(uintptr_t)gaddr)
               : "memory");
}
__device__ inline void wait_async0() {
  asm volatile("s_wait_asynccnt 0" ::: "memory");
}

__global__ __launch_bounds__(256) void gemm_glu_kernel(const __bf16* __restrict__ A,
                                                       const __bf16* __restrict__ Wb,
                                                       const float* __restrict__ bias,
                                                       float* __restrict__ out) {
  __shared__ __bf16 As[2][BM * LDSW];   // double-buffered A tile
  __shared__ __bf16 Bs[2][BM * LDSW];   // rows 0..63: a-cols, 64..127: g-cols

  const int tid  = threadIdx.x;
  const int m0   = blockIdx.x * BM;
  const int n0   = blockIdx.y * BN;
  const int wid  = tid >> 5;
  const int lane = tid & 31;
  const int wm   = wid & 3;          // wave M index (0..3) -> 32 rows each
  const int wn   = wid >> 2;         // wave N index (0..1) -> 32 out cols each
  const int frow = lane & 15;
  const int fkA  = (lane >> 4) << 4; // A: contiguous 16 K per half-wave
  const int fkB  = (lane >> 4) << 4; // B: contiguous 16 K per half-wave
  const int mhalf = (lane >> 4) * 8;

  // staging assignment: thread -> (row, 8-elt column chunk)
  const int sr = tid >> 2;           // 0..63
  const int sc = (tid & 3) * 8;      // 0,8,16,24 bf16 elements

  // per-thread global source pointers (advance by kc)
  const __bf16* gA0 = &A[(size_t)(m0 + sr) * GK + sc];
  const __bf16* gA1 = &A[(size_t)(m0 + sr + 64) * GK + sc];
  const __bf16* gB0 = &Wb[(size_t)(n0 + sr) * GK + sc];
  const __bf16* gB1 = &Wb[(size_t)(GN + n0 + sr) * GK + sc];

  // per-thread LDS destination offsets (flat addr low 32 bits == LDS offset)
  const int si0 = sr * LDSW + sc;
  const int si1 = (sr + 64) * LDSW + sc;

  v8f acc_a[2][2], acc_g[2][2];
#pragma unroll
  for (int tm = 0; tm < 2; ++tm)
#pragma unroll
    for (int tn = 0; tn < 2; ++tn) { acc_a[tm][tn] = vzero8(); acc_g[tm][tn] = vzero8(); }

  // prologue: issue async stage of first K-chunk into buffer 0
  async_b128((uint32_t)(uintptr_t)&As[0][si0], gA0);
  async_b128((uint32_t)(uintptr_t)&As[0][si1], gA1);
  async_b128((uint32_t)(uintptr_t)&Bs[0][si0], gB0);
  async_b128((uint32_t)(uintptr_t)&Bs[0][si1], gB1);

  for (int kc = 0; kc < GK; kc += KC) {
    const int cur = (kc / KC) & 1;
    const int nxt = cur ^ 1;

    wait_async0();        // my async writes (issued last iter) have landed
    __syncthreads();      // everyone's writes have landed; prior reads done

    if (kc + KC < GK) {   // issue next chunk into the other buffer
      const int kn = kc + KC;
      async_b128((uint32_t)(uintptr_t)&As[nxt][si0], gA0 + kn);
      async_b128((uint32_t)(uintptr_t)&As[nxt][si1], gA1 + kn);
      async_b128((uint32_t)(uintptr_t)&Bs[nxt][si0], gB0 + kn);
      async_b128((uint32_t)(uintptr_t)&Bs[nxt][si1], gB1 + kn);
    }

    // ---- fragments from current buffer ----
    FragB16 af[2], baf[2], bgf[2];
#pragma unroll
    for (int tm = 0; tm < 2; ++tm) {
      const int r = wm * 32 + tm * 16 + frow;
      af[tm].p.lo = *(const v8bf*)&As[cur][r * LDSW + fkA];
      af[tm].p.hi = *(const v8bf*)&As[cur][r * LDSW + fkA + 8];
    }
#pragma unroll
    for (int tn = 0; tn < 2; ++tn) {
      const int r = wn * 32 + tn * 16 + frow;
      baf[tn].p.lo = *(const v8bf*)&Bs[cur][r * LDSW + fkB];
      baf[tn].p.hi = *(const v8bf*)&Bs[cur][r * LDSW + fkB + 8];
      bgf[tn].p.lo = *(const v8bf*)&Bs[cur][(r + 64) * LDSW + fkB];
      bgf[tn].p.hi = *(const v8bf*)&Bs[cur][(r + 64) * LDSW + fkB + 8];
    }

    // ---- 8 WMMAs per K step ----
#pragma unroll
    for (int tm = 0; tm < 2; ++tm)
#pragma unroll
      for (int tn = 0; tn < 2; ++tn) {
        acc_a[tm][tn] = __builtin_amdgcn_wmma_f32_16x16x32_bf16(
            false, af[tm].v, false, baf[tn].v, (short)0, acc_a[tm][tn], false, false);
        acc_g[tm][tn] = __builtin_amdgcn_wmma_f32_16x16x32_bf16(
            false, af[tm].v, false, bgf[tn].v, (short)0, acc_g[tm][tn], false, false);
      }
  }

  // ---- epilogue: bias + GLU, fp32 store ----
#pragma unroll
  for (int tm = 0; tm < 2; ++tm)
#pragma unroll
    for (int tn = 0; tn < 2; ++tn) {
      const int ncol = n0 + wn * 32 + tn * 16 + frow;
      const float bA = bias[ncol];
      const float bG = bias[GN + ncol];
      const int mbase = m0 + wm * 32 + tm * 16 + mhalf;
#pragma unroll
      for (int i = 0; i < 8; ++i) {
        float za = acc_a[tm][tn][i] + bA;
        float zg = acc_g[tm][tn][i] + bG;
        float s  = 1.0f / (1.0f + expf(-zg));
        out[(size_t)(mbase + i) * GN + ncol] = za * s;
      }
    }
}

// =====================================================================
// launch
// =====================================================================
extern "C" void kernel_launch(void* const* d_in, const int* in_sizes, int n_in,
                              void* d_out, int out_size, void* d_ws, size_t ws_size,
                              hipStream_t stream) {
  const float* u   = (const float*)d_in[0];   // [B, L, H]
  const float* ker = (const float*)d_in[1];   // [1, H, Lk]
  const float* D   = (const float*)d_in[2];   // [1, H]
  const float* W   = (const float*)d_in[3];   // [2H, H]
  const float* b   = (const float*)d_in[4];   // [2H]
  float* out = (float*)d_out;                 // [B, L, H]

  // workspace layout
  char* ws = (char*)d_ws;
  __bf16* Yb  = (__bf16*)ws;                                    // GM*GK bf16  = 32 MiB
  __bf16* Wb  = (__bf16*)(ws + (size_t)GM * GK * 2);            // 2*GN*GK bf16 = 16 MiB
  float*  ksq = (float*)(ws + (size_t)GM * GK * 2 + (size_t)2 * GN * GK * 2); // 1 MiB

  // 1) squash conv kernel
  {
    int n = CH * CLK;
    squash_kernel<<<(n + 255) / 256, 256, 0, stream>>>(ker, ksq, n);
  }
  // 2) W -> bf16
  {
    int n = 2 * GN * GK;
    w2bf_kernel<<<(n + 255) / 256, 256, 0, stream>>>(W, Wb, n);
  }
  // 3) conv + skip + GELU -> bf16 activations
  {
    dim3 grid(CH / 256, CL / TL, CB);
    conv_gelu_kernel<<<grid, 256, 0, stream>>>(u, ksq, D, Yb);
  }
  // 4) WMMA GEMM + bias + GLU (async LDS staging, double buffered)
  {
    dim3 grid(GM / BM, GN / BN);
    gemm_glu_kernel<<<grid, 256, 0, stream>>>(Yb, Wb, b, out);
  }
}
